// MOE_56831007261004
// MI455X (gfx1250) — matrix operand order repaired and checked by reading, not compiled, not run on hardware
//
#include <hip/hip_runtime.h>
#include <stdint.h>

// Problem constants (B=2,T=2048 -> NT tokens)
#define NT 4096
#define DD 1024
#define FF 4096
#define NE 8

typedef __attribute__((ext_vector_type(16))) __bf16 v16bf;
typedef __attribute__((ext_vector_type(8)))  float  v8f;
typedef __attribute__((ext_vector_type(4)))  int    v4i_;

#define GLOBAL_AS __attribute__((address_space(1)))
#define LDS_AS    __attribute__((address_space(3)))

union FragU {            // 32 bytes: two b128 LDS loads -> one WMMA operand
  uint4 q[2];
  v16bf v;
};

__device__ __forceinline__ unsigned int f2bf(float f) {
  union { float f; unsigned int u; } c; c.f = f;
  unsigned int u = c.u;
  u += 0x7FFFu + ((u >> 16) & 1u);   // round-to-nearest-even
  return u >> 16;
}
__device__ __forceinline__ unsigned int pk2bf(float a, float b) {
  return f2bf(a) | (f2bf(b) << 16);
}

// branch-free tanh-approx gelu: tanh(u) = 1 - 2/(exp(2u)+1)
__device__ __forceinline__ float gelu_fast(float v) {
  float t2 = 1.5957691216057308f * (v + 0.044715f * v * v * v);  // 2*u
  float e  = __expf(t2);
  float th = 1.f - 2.f / (e + 1.f);
  return 0.5f * v * (1.f + th);
}

// ---- async global->LDS (gfx1250), guarded fallback to plain copy ----------
#if __has_builtin(__builtin_amdgcn_global_load_async_to_lds_b128)
#define USE_ASYNC_LDS 1
__device__ __forceinline__ void cp16_async(void* lds, const void* g) {
  __builtin_amdgcn_global_load_async_to_lds_b128(
      (GLOBAL_AS v4i_*)(uintptr_t)g,
      (LDS_AS v4i_*)(unsigned int)(uintptr_t)lds, 0, 0);
}
__device__ __forceinline__ void async_wait() {
#if __has_builtin(__builtin_amdgcn_s_wait_asynccnt)
  __builtin_amdgcn_s_wait_asynccnt(0);
#else
  asm volatile("s_wait_asynccnt 0x0" ::: "memory");
#endif
}
#else
#define USE_ASYNC_LDS 0
__device__ __forceinline__ void cp16_async(void* lds, const void* g) {
  *(uint4*)lds = *(const uint4*)g;
}
__device__ __forceinline__ void async_wait() {}
#endif

// ---------------------------------------------------------------- gating ----
__global__ __launch_bounds__(256) void moe_gate_kernel(
    const float* __restrict__ x, const float* __restrict__ Wg,
    int* counts, int* list, float* gates) {
  const int wave = threadIdx.x >> 5;
  const int lane = threadIdx.x & 31;
  const int t = blockIdx.x * 8 + wave;

  float acc[NE];
#pragma unroll
  for (int e = 0; e < NE; ++e) acc[e] = 0.f;
  const float* xr = x + (size_t)t * DD;
  for (int d = lane; d < DD; d += 32) {
    float xv = xr[d];
#pragma unroll
    for (int e = 0; e < NE; ++e) acc[e] += xv * Wg[d * NE + e];
  }
#pragma unroll
  for (int m = 16; m >= 1; m >>= 1) {
#pragma unroll
    for (int e = 0; e < NE; ++e) acc[e] += __shfl_xor(acc[e], m, 32);
  }
  if (lane == 0) {
    int e0 = 0; float g0 = acc[0];
#pragma unroll
    for (int e = 1; e < NE; ++e) if (acc[e] > g0) { g0 = acc[e]; e0 = e; }  // strict >: first index on ties
    int e1 = (e0 == 0) ? 1 : 0; float g1 = acc[e1];
#pragma unroll
    for (int e = 0; e < NE; ++e) if (e != e0 && acc[e] > g1) { g1 = acc[e]; e1 = e; }
    float r  = __expf(g1 - g0);
    float w0 = 1.f / (1.f + r);
    float w1 = r * w0;
    int s0 = atomicAdd(&counts[e0], 1);
    list[e0 * NT + s0] = t; gates[e0 * NT + s0] = w0;
    int s1 = atomicAdd(&counts[e1], 1);
    list[e1 * NT + s1] = t; gates[e1 * NT + s1] = w1;
  }
}

__global__ void moe_scan_kernel(const int* counts, int* offsets) {
  if (threadIdx.x == 0) {
    int o = 0;
    for (int e = 0; e < NE; ++e) { offsets[e] = o; o += counts[e]; }
    offsets[NE] = o;
  }
}

// ------------------------------------------------------------- x -> bf16 ----
__global__ __launch_bounds__(256) void moe_cvt_x_kernel(
    const float* __restrict__ x, unsigned short* __restrict__ Xb) {
  const int g = (blockIdx.x * 256 + threadIdx.x) * 4;
  float4 v = *(const float4*)(x + g);
  uint2 o;
  o.x = pk2bf(v.x, v.y);
  o.y = pk2bf(v.z, v.w);
  *(uint2*)(Xb + g) = o;
}

// --------------------------------------------------------- shared pieces ----
// B stager: thread t handles a 4(k) x 4(n) micro-tile of the fp32 weight
// chunk [k0..k0+31] x [base..base+127]; 4 packed b64 LDS stores, n-major.
__device__ __forceinline__ void stage_B_fp32(
    unsigned short* lsB, const float* __restrict__ W, int ld, int base,
    int k0, int tid) {
  const int kq = tid >> 5;          // 0..7 -> k = k0 + kq*4
  const int fi = tid & 31;          // n = base + fi*4 .. +3
  union F4 { float4 v; float a[4]; } r[4];
#pragma unroll
  for (int j = 0; j < 4; ++j)
    r[j].v = *(const float4*)(W + (size_t)(k0 + kq * 4 + j) * ld + base + fi * 4);
#pragma unroll
  for (int c = 0; c < 4; ++c) {
    uint2 o;
    o.x = pk2bf(r[0].a[c], r[1].a[c]);   // k, k+1
    o.y = pk2bf(r[2].a[c], r[3].a[c]);   // k+2, k+3
    *(uint2*)&lsB[(fi * 4 + c) * 40 + kq * 4] = o;
  }
}

// A stager: copy 16 contiguous bf16 (two async b128) into row-major lsA
__device__ __forceinline__ void stage_A_bf16(
    unsigned short* lsA, const unsigned short* __restrict__ src,
    int ar, int ah) {
  unsigned short* d = &lsA[ar * 40 + ah * 16];
  cp16_async(d, src);
  cp16_async(d + 8, src + 8);
}

// zero rows [tail..127] of an lsA buffer (only the K region, 32 ushorts/row)
__device__ __forceinline__ void zero_tail_rows(unsigned short* lsA, int tail, int tid) {
  if (tail < 128) {
    for (int i = tid; i < (128 - tail) * 4; i += 256) {
      int r = tail + (i >> 2);
      *(uint4*)&lsA[r * 40 + (i & 3) * 8] = make_uint4(0, 0, 0, 0);
    }
  }
}

// one K=32 step: load fragments + 8 WMMAs (wave tile 64x32 of a 128x128 block)
__device__ __forceinline__ void mma_step(
    const unsigned short* lsA, const unsigned short* lsB,
    v8f (&acc)[4][2], int wm, int wn, int lo, int hi) {
  FragU a[4], b[2];
#pragma unroll
  for (int mi = 0; mi < 4; ++mi) {
    int row = wm * 64 + mi * 16 + lo;
    a[mi].q[0] = *(const uint4*)&lsA[row * 40 +      hi * 8];
    a[mi].q[1] = *(const uint4*)&lsA[row * 40 + 16 + hi * 8];
  }
#pragma unroll
  for (int ni = 0; ni < 2; ++ni) {
    int f = wn * 32 + ni * 16 + lo;
    b[ni].q[0] = *(const uint4*)&lsB[f * 40 +      hi * 8];
    b[ni].q[1] = *(const uint4*)&lsB[f * 40 + 16 + hi * 8];
  }
#pragma unroll
  for (int mi = 0; mi < 4; ++mi)
#pragma unroll
    for (int ni = 0; ni < 2; ++ni)
      acc[mi][ni] = __builtin_amdgcn_wmma_f32_16x16x32_bf16(
          false, a[mi].v, false, b[ni].v, (short)0, acc[mi][ni], false, false);
}

// --------------------------------------------------------------- GEMM 1 -----
// H[off_e + slot, f] = gelu( X[tok] . W1_e[:, f] + b1_e[f] ), bf16 output
__global__ __launch_bounds__(256) void moe_gemm1_kernel(
    const unsigned short* __restrict__ Xb, const float* __restrict__ W1,
    const float* __restrict__ b1,
    const int* __restrict__ counts, const int* __restrict__ offsets,
    const int* __restrict__ list, unsigned short* __restrict__ H) {
  const int e   = blockIdx.z;
  const int cnt = counts[e];
  const int mt  = blockIdx.y;
  if (mt * 128 >= cnt) return;                 // uniform early exit
  const int fb  = blockIdx.x * 128;
  const int off = offsets[e];
  const float* W1e = W1 + (size_t)e * DD * FF;

  __shared__ __align__(16) unsigned short lsA[2][128 * 40];  // double buffered
  __shared__ __align__(16) unsigned short lsB[2][128 * 40];

  const int tid  = threadIdx.x;
  const int lane = tid & 31;
  const int wid  = tid >> 5;
  const int wm = wid & 1, wn = wid >> 1;       // 2x4 wave grid over 128x128
  const int lo = lane & 15, hi = lane >> 4;

  v8f acc[4][2];
#pragma unroll
  for (int mi = 0; mi < 4; ++mi)
#pragma unroll
    for (int ni = 0; ni < 2; ++ni)
#pragma unroll
      for (int r = 0; r < 8; ++r) acc[mi][ni][r] = 0.f;

  const int ar = tid >> 1, ah = tid & 1;               // A stager: row, k-half
  const int aslot  = mt * 128 + ar;
  const int avalid = (aslot < cnt);
  const size_t axrow =
      avalid ? (size_t)list[e * NT + aslot] * DD : 0;    // gathered token row

  zero_tail_rows(lsA[0], cnt - mt * 128, tid);           // persists: never rewritten
  zero_tail_rows(lsA[1], cnt - mt * 128, tid);

  // prologue: stage chunk 0 into buffer 0
  if (avalid) stage_A_bf16(lsA[0], Xb + axrow + ah * 16, ar, ah);
  stage_B_fp32(lsB[0], W1e, FF, fb, 0, tid);

  const int NK = DD / 32;
  for (int ki = 0; ki < NK; ++ki) {
    int cur = ki & 1, nxt = cur ^ 1;
    async_wait();          // previous-iter asyncs into lsA[cur] complete
    __syncthreads();       // + everyone's DS stores into buffers flushed
    if (ki + 1 < NK) {     // stage next chunk (overlaps with compute below)
      int k0 = (ki + 1) * 32;
      if (avalid) stage_A_bf16(lsA[nxt], Xb + axrow + k0 + ah * 16, ar, ah);
      stage_B_fp32(lsB[nxt], W1e, FF, fb, k0, tid);
    }
    mma_step(lsA[cur], lsB[cur], acc, wm, wn, lo, hi);
  }

  const float* b1e = b1 + (size_t)e * FF;
#pragma unroll
  for (int mi = 0; mi < 4; ++mi)
#pragma unroll
    for (int r = 0; r < 8; ++r) {
      int lrow = wm * 64 + mi * 16 + r + 8 * hi;
      int slot = mt * 128 + lrow;
      if (slot < cnt) {
        size_t hrow = (size_t)(off + slot) * FF;
#pragma unroll
        for (int ni = 0; ni < 2; ++ni) {
          int col = fb + wn * 32 + ni * 16 + lo;
          float v = gelu_fast(acc[mi][ni][r] + b1e[col]);
          H[hrow + col] = (unsigned short)f2bf(v);
        }
      }
    }
}

// --------------------------------------------------------------- GEMM 2 -----
// out[tok] += gate * ( H[row] . W2_e[:, d] + b2_e[d] )
__global__ __launch_bounds__(256) void moe_gemm2_kernel(
    const unsigned short* __restrict__ H, const float* __restrict__ W2,
    const float* __restrict__ b2,
    const int* __restrict__ counts, const int* __restrict__ offsets,
    const int* __restrict__ list, const float* __restrict__ gates,
    float* __restrict__ out) {
  const int e   = blockIdx.z;
  const int cnt = counts[e];
  const int mt  = blockIdx.y;
  if (mt * 128 >= cnt) return;
  const int nb  = blockIdx.x * 128;
  const int off = offsets[e];
  const float* W2e = W2 + (size_t)e * FF * DD;

  __shared__ __align__(16) unsigned short lsA[2][128 * 40];
  __shared__ __align__(16) unsigned short lsB[2][128 * 40];

  const int tid  = threadIdx.x;
  const int lane = tid & 31;
  const int wid  = tid >> 5;
  const int wm = wid & 1, wn = wid >> 1;
  const int lo = lane & 15, hi = lane >> 4;

  v8f acc[4][2];
#pragma unroll
  for (int mi = 0; mi < 4; ++mi)
#pragma unroll
    for (int ni = 0; ni < 2; ++ni)
#pragma unroll
      for (int r = 0; r < 8; ++r) acc[mi][ni][r] = 0.f;

  const int ar = tid >> 1, ah = tid & 1;
  const int aslot  = mt * 128 + ar;
  const int avalid = (aslot < cnt);
  const size_t ahrow = avalid ? (size_t)(off + aslot) * FF : 0;

  zero_tail_rows(lsA[0], cnt - mt * 128, tid);
  zero_tail_rows(lsA[1], cnt - mt * 128, tid);

  if (avalid) stage_A_bf16(lsA[0], H + ahrow + ah * 16, ar, ah);
  stage_B_fp32(lsB[0], W2e, DD, nb, 0, tid);

  const int NK = FF / 32;
  for (int ki = 0; ki < NK; ++ki) {
    int cur = ki & 1, nxt = cur ^ 1;
    async_wait();
    __syncthreads();
    if (ki + 1 < NK) {
      int k0 = (ki + 1) * 32;
      if (avalid) stage_A_bf16(lsA[nxt], H + ahrow + k0 + ah * 16, ar, ah);
      stage_B_fp32(lsB[nxt], W2e, DD, nb, k0, tid);
    }
    mma_step(lsA[cur], lsB[cur], acc, wm, wn, lo, hi);
  }

  const float* b2e = b2 + (size_t)e * DD;
#pragma unroll
  for (int mi = 0; mi < 4; ++mi)
#pragma unroll
    for (int r = 0; r < 8; ++r) {
      int lrow = wm * 64 + mi * 16 + r + 8 * hi;
      int slot = mt * 128 + lrow;
      if (slot < cnt) {
        int tok  = list[e * NT + slot];
        float g  = gates[e * NT + slot];
#pragma unroll
        for (int ni = 0; ni < 2; ++ni) {
          int col = nb + wn * 32 + ni * 16 + lo;
          float v = (acc[mi][ni][r] + b2e[col]) * g;
          atomicAdd(&out[(size_t)tok * DD + col], v);
        }
      }
    }
}

// --------------------------------------------------------------- launch -----
extern "C" void kernel_launch(void* const* d_in, const int* in_sizes, int n_in,
                              void* d_out, int out_size, void* d_ws, size_t ws_size,
                              hipStream_t stream) {
  const float* x  = (const float*)d_in[0];
  const float* Wg = (const float*)d_in[1];
  const float* W1 = (const float*)d_in[2];
  const float* b1 = (const float*)d_in[3];
  const float* W2 = (const float*)d_in[4];
  const float* b2 = (const float*)d_in[5];
  float* out = (float*)d_out;

  char* ws = (char*)d_ws;
  int*   counts  = (int*)ws;                       // 8 ints
  int*   offsets = (int*)(ws + 64);                // 9 ints
  int*   list    = (int*)(ws + 256);               // E*NT ints   (128 KB)
  float* gates   = (float*)(ws + 256 + (size_t)NE * NT * 4);  // (128 KB)
  unsigned short* Xb = (unsigned short*)(ws + (1u << 20));    // NT*D bf16 = 8 MB
  unsigned short* H  = (unsigned short*)(ws + (1u << 20) + (size_t)NT * DD * 2); // 64 MB

  (void)hipMemsetAsync(ws, 0, 256, stream);
  (void)hipMemsetAsync(out, 0, (size_t)out_size * sizeof(float), stream);

  moe_cvt_x_kernel<<<(NT * DD) / 1024, 256, 0, stream>>>(x, Xb);
  moe_gate_kernel<<<NT / 8, 256, 0, stream>>>(x, Wg, counts, list, gates);
  moe_scan_kernel<<<1, 32, 0, stream>>>(counts, offsets);
  moe_gemm1_kernel<<<dim3(FF / 128, NT / 128, NE), 256, 0, stream>>>(
      Xb, W1, b1, counts, offsets, list, H);
  moe_gemm2_kernel<<<dim3(DD / 128, NT / 128, NE), 256, 0, stream>>>(
      H, W2, b2, counts, offsets, list, gates, out);
}
